// MyModel_33277406609696
// MI455X (gfx1250) — matrix operand order repaired and checked by reading, not compiled, not run on hardware
//
#include <hip/hip_runtime.h>

typedef _Float16 f16;
typedef __attribute__((ext_vector_type(16))) _Float16 v16h;
typedef __attribute__((ext_vector_type(8)))  float    v8f;

#define NN 50000
#define NE 1600000
#define FIN 64
#define C 128

// ---------------- device helpers ----------------

__device__ __forceinline__ float silu_f(float x) { return x / (1.0f + __expf(-x)); }
__device__ __forceinline__ float gelu_f(float x) {
  return 0.5f * x * (1.0f + erff(x * 0.7071067811865475f));
}

// A operand (16x32 f16) from LDS row-major tile, pitch lda_h (f16 units).
// Layout per CDNA5 ISA 7.12.2: lane m=lane&15 holds row M=m; lanes 0-15 get
// K 0..7 (VGPR0-3) and 16..23 (VGPR4-7); lanes 16-31 get K 8..15 / 24..31.
__device__ __forceinline__ v16h load_a(const f16* ldsA, int lda_h, int lane, int kt) {
  const int m = lane & 15;
  const int h4 = (lane >> 4) * 4;                       // dwords
  const unsigned int* p = (const unsigned int*)(ldsA + m * lda_h + kt * 32);
  union { v16h h; unsigned int u[8]; } r;
#pragma unroll
  for (int v = 0; v < 8; ++v)
    r.u[v] = p[((v & 4) ? 8 : 0) + h4 + (v & 3)];
  return r.h;
}

// B operand from pre-packed weights: tile (kt,nt), per-lane 16 contiguous f16.
__device__ __forceinline__ v16h load_b(const f16* Bp, int kt, int nt, int lane) {
  return *(const v16h*)(Bp + (((((kt << 3) + nt) << 5) + lane) << 4));
}

template<int KT>
__device__ __forceinline__ v8f gemm_nt(const f16* ldsA, int lda_h, const f16* Bp,
                                       int nt, int lane) {
  v8f acc = {};
#pragma unroll
  for (int kt = 0; kt < KT; ++kt) {
    v16h a = load_a(ldsA, lda_h, lane, kt);
    v16h b = load_b(Bp, kt, nt, lane);
    acc = __builtin_amdgcn_wmma_f32_16x16x32_f16(
        false, a, false, b, (short)0, acc, false, false);
  }
  return acc;
}

// Two M-tiles sharing one B stream: each loaded B fragment feeds two WMMAs,
// halving L2 weight traffic for the edge pipeline.
template<int KT>
__device__ __forceinline__ void gemm_nt2(const f16* A0, const f16* A1, int lda_h,
                                         const f16* Bp, int nt, int lane,
                                         v8f& acc0, v8f& acc1) {
#pragma unroll
  for (int kt = 0; kt < KT; ++kt) {
    v16h b = load_b(Bp, kt, nt, lane);
    v16h a0 = load_a(A0, lda_h, lane, kt);
    v16h a1 = load_a(A1, lda_h, lane, kt);
    acc0 = __builtin_amdgcn_wmma_f32_16x16x32_f16(
        false, a0, false, b, (short)0, acc0, false, false);
    acc1 = __builtin_amdgcn_wmma_f32_16x16x32_f16(
        false, a1, false, b, (short)0, acc1, false, false);
  }
}

// ---------------- weight pre-pack ----------------
// Packs row-major f32 [K,128] into per-(kt,nt,lane) contiguous 16-f16 groups
// matching the B layout assumed by load_b: k = kt*32 + (e>>3)*16 + (lane>>4)*8 + (e&7),
// n = nt*16 + (lane&15), element e = 0..15.
__global__ void pack_b(const float* __restrict__ src, f16* __restrict__ dst, int K) {
  int idx = blockIdx.x * 256 + threadIdx.x;
  if (idx >= K * C) return;
  int e    = idx & 15;
  int lane = (idx >> 4) & 31;
  int tnum = idx >> 9;
  int nt = tnum & 7;
  int kt = tnum >> 3;
  int k = kt * 32 + (e >> 3) * 16 + (lane >> 4) * 8 + (e & 7);
  int n = nt * 16 + (lane & 15);
  dst[idx] = (f16)src[k * C + n];
}

// ---------------- embedding ----------------
__global__ __launch_bounds__(32) void embed_kernel(
    const float* __restrict__ attr,
    const f16* __restrict__ w1p, const float* __restrict__ b1,
    const float* __restrict__ g1, const float* __restrict__ be1,
    const f16* __restrict__ w2p, const float* __restrict__ b2,
    const float* __restrict__ g2, const float* __restrict__ be2,
    float* __restrict__ h, f16* __restrict__ h16) {
  __shared__ __align__(16) f16 sA[16 * 72];
  __shared__ __align__(16) f16 sB[16 * 136];
  __shared__ float sH[16 * 132];
  const int lane = threadIdx.x;
  const int m = lane & 15, half = lane >> 4;
  const long node = (long)blockIdx.x * 16 + m;

  { // load node_attr [16x64] -> f16 tile
    const float4* srow = (const float4*)(attr + node * FIN + half * 32);
    f16* drow = sA + m * 72 + half * 32;
#pragma unroll
    for (int i = 0; i < 8; ++i) {
      float4 f = srow[i];
      drow[4*i+0] = (f16)f.x; drow[4*i+1] = (f16)f.y;
      drow[4*i+2] = (f16)f.z; drow[4*i+3] = (f16)f.w;
    }
  }
  __syncthreads();

  for (int nt = 0; nt < 8; ++nt) {
    v8f acc = gemm_nt<2>(sA, 72, w1p, nt, lane);
    int col = nt * 16 + m; float bv = b1[col];
#pragma unroll
    for (int v = 0; v < 8; ++v) sH[(half*8+v)*132 + col] = acc[v] + bv;
  }
  __syncthreads();

  if (half == 0) { // LN(ln1) then gelu -> f16 for next GEMM
    float* row = sH + m * 132;
    float s = 0.f, ss = 0.f;
    for (int k = 0; k < C; ++k) { float t = row[k]; s += t; ss += t * t; }
    float mu = s * (1.0f / C);
    float inv = rsqrtf(ss * (1.0f / C) - mu * mu + 1e-5f);
    f16* brow = sB + m * 136;
    for (int k = 0; k < C; ++k)
      brow[k] = (f16)gelu_f((row[k] - mu) * inv * g1[k] + be1[k]);
  }
  __syncthreads();

  for (int nt = 0; nt < 8; ++nt) {
    v8f acc = gemm_nt<4>(sB, 136, w2p, nt, lane);
    int col = nt * 16 + m; float bv = b2[col];
#pragma unroll
    for (int v = 0; v < 8; ++v) sH[(half*8+v)*132 + col] = acc[v] + bv;
  }
  __syncthreads();

  if (half == 0) { // LN(ln2) then gelu -> h, h16
    float* row = sH + m * 132;
    float s = 0.f, ss = 0.f;
    for (int k = 0; k < C; ++k) { float t = row[k]; s += t; ss += t * t; }
    float mu = s * (1.0f / C);
    float inv = rsqrtf(ss * (1.0f / C) - mu * mu + 1e-5f);
    for (int k = 0; k < C; ++k) {
      float o = gelu_f((row[k] - mu) * inv * g2[k] + be2[k]);
      h[node * C + k] = o;
      h16[node * C + k] = (f16)o;
    }
  }
}

// ---------------- fused edge pipeline ----------------
// One wave per 32-edge tile (2 x 16-row WMMA tiles sharing each B fragment):
// gather -> GEMM1(K=257) -> GEMM2(+scatter h_agg) -> GEMM3 -> coord head.
__global__ __launch_bounds__(32) void edge_kernel(
    const int* __restrict__ src, const int* __restrict__ dst,
    const f16* __restrict__ h16, const float* __restrict__ xpos,
    const f16* __restrict__ ew1p, const float* __restrict__ ew1_last,
    const float* __restrict__ eb1,
    const f16* __restrict__ ew2p, const float* __restrict__ eb2,
    const f16* __restrict__ cw1p, const float* __restrict__ cb1,
    const float* __restrict__ cw2,
    float* __restrict__ h_agg, float* __restrict__ x_agg, int do_coord) {
  __shared__ __align__(16) f16 sF[2 * 16 * 264];  // f tiles, then m tiles (pitch 136)
  __shared__ __align__(16) f16 sM[2 * 16 * 136];  // m1 tiles, then t tiles
  __shared__ float sRad[32];
  __shared__ int   sIdx[64];                      // src[0..31], dst[32..63]
  const int lane = threadIdx.x;
  const int m = lane & 15, half = lane >> 4;
  const long e0 = (long)blockIdx.x * 32;

  // each lane owns one edge: indices, radial, normalized diff (kept in regs)
  const int sN = src[e0 + lane], dN = dst[e0 + lane];
  sIdx[lane] = sN; sIdx[32 + lane] = dN;
  float dx = xpos[sN*3+0] - xpos[dN*3+0];
  float dy = xpos[sN*3+1] - xpos[dN*3+1];
  float dz = xpos[sN*3+2] - xpos[dN*3+2];
  float rad = dx*dx + dy*dy + dz*dz;
  sRad[lane] = rad;
  float rinv = 1.0f / (sqrtf(rad) + 1e-30f);
  dx *= rinv; dy *= rinv; dz *= rinv;

  { // gather: lane writes its edge's rows into tile t=lane>>4, row m
    f16* frow = sF + half * (16 * 264) + m * 264;
    const uint4* srow = (const uint4*)(h16 + (long)sN * C);
    const uint4* drow = (const uint4*)(h16 + (long)dN * C);
    uint4* fs = (uint4*)frow;
    uint4* fd = (uint4*)(frow + 128);
#pragma unroll
    for (int i = 0; i < 8; ++i) { fs[i] = srow[i]; fd[i] = drow[i]; }
  }
  __syncthreads();

  // GEMM1: m1 = silu(f @ ew1 + eb1); radial column (k=256) as VALU rank-1
  for (int nt = 0; nt < 8; ++nt) {
    v8f a0 = {}, a1 = {};
    gemm_nt2<8>(sF, sF + 16 * 264, 264, ew1p, nt, lane, a0, a1);
    int col = nt * 16 + m;
    float wl = ew1_last[col], bv = eb1[col];
#pragma unroll
    for (int v = 0; v < 8; ++v) {
      int row = half * 8 + v;
      sM[row * 136 + col] = (f16)silu_f(a0[v] + sRad[row] * wl + bv);
      sM[(16 + row) * 136 + col] = (f16)silu_f(a1[v] + sRad[16 + row] * wl + bv);
    }
  }
  __syncthreads();

  // GEMM2: msg = silu(m1 @ ew2 + eb2); scatter-add h_agg; keep msg in sF
  for (int nt = 0; nt < 8; ++nt) {
    v8f a0 = {}, a1 = {};
    gemm_nt2<4>(sM, sM + 16 * 136, 136, ew2p, nt, lane, a0, a1);
    int col = nt * 16 + m;
    float bv = eb2[col];
#pragma unroll
    for (int v = 0; v < 8; ++v) {
      int row = half * 8 + v;
      float v0 = silu_f(a0[v] + bv);
      float v1 = silu_f(a1[v] + bv);
      sF[row * 136 + col] = (f16)v0;
      sF[(16 + row) * 136 + col] = (f16)v1;
      atomicAdd(&h_agg[(long)sIdx[32 + row] * C + col], v0);
      atomicAdd(&h_agg[(long)sIdx[48 + row] * C + col], v1);
    }
  }
  __syncthreads();

  // GEMM3: t = silu(msg @ cw1 + cb1) -> sM
  for (int nt = 0; nt < 8; ++nt) {
    v8f a0 = {}, a1 = {};
    gemm_nt2<4>(sF, sF + 16 * 136, 136, cw1p, nt, lane, a0, a1);
    int col = nt * 16 + m;
    float bv = cb1[col];
#pragma unroll
    for (int v = 0; v < 8; ++v) {
      sM[(half*8+v) * 136 + col] = (f16)silu_f(a0[v] + bv);
      sM[(16 + half*8+v) * 136 + col] = (f16)silu_f(a1[v] + bv);
    }
  }
  __syncthreads();

  if (do_coord) { // cm = t @ cw2: each lane owns its edge's full 128-dot
    const f16* trow = sM + half * (16 * 136) + m * 136;
    float cm = 0.f;
#pragma unroll 8
    for (int k = 0; k < C; ++k) cm += (float)trow[k] * cw2[k];
    atomicAdd(&x_agg[dN*3+0], cm * dx);
    atomicAdd(&x_agg[dN*3+1], cm * dy);
    atomicAdd(&x_agg[dN*3+2], cm * dz);
  }
}

// ---------------- node update ----------------
__global__ __launch_bounds__(32) void node_kernel(
    float* __restrict__ h, f16* __restrict__ h16,
    const float* __restrict__ h_agg,
    const f16* __restrict__ nw1p, const float* __restrict__ nb1,
    const f16* __restrict__ nw2p, const float* __restrict__ nb2,
    const float* __restrict__ ln_g, const float* __restrict__ ln_b,
    float* __restrict__ coord, const float* __restrict__ x_agg,
    int post_ln, int do_coord) {
  __shared__ __align__(16) f16 sA[16 * 264];
  __shared__ __align__(16) f16 sB[16 * 136];
  __shared__ float sH[16 * 132];
  const int lane = threadIdx.x;
  const int m = lane & 15, half = lane >> 4;
  const long node = (long)blockIdx.x * 16 + m;

  if (half == 0) { // concat: h (already f16)
    const uint4* srow = (const uint4*)(h16 + node * C);
    uint4* drow = (uint4*)(sA + m * 264);
#pragma unroll
    for (int i = 0; i < 8; ++i) drow[i] = srow[i];
  } else {         // concat: h_agg f32 -> f16
    const float4* srow = (const float4*)(h_agg + node * C);
    f16* drow = sA + m * 264 + 128;
#pragma unroll
    for (int i = 0; i < 32; ++i) {
      float4 f = srow[i];
      drow[4*i+0] = (f16)f.x; drow[4*i+1] = (f16)f.y;
      drow[4*i+2] = (f16)f.z; drow[4*i+3] = (f16)f.w;
    }
  }
  __syncthreads();

  for (int nt = 0; nt < 8; ++nt) {
    v8f acc = gemm_nt<8>(sA, 264, nw1p, nt, lane);
    int col = nt * 16 + m; float bv = nb1[col];
#pragma unroll
    for (int v = 0; v < 8; ++v)
      sB[(half*8+v) * 136 + col] = (f16)silu_f(acc[v] + bv);
  }
  __syncthreads();

  for (int nt = 0; nt < 8; ++nt) {
    v8f acc = gemm_nt<4>(sB, 136, nw2p, nt, lane);
    int col = nt * 16 + m; float bv = nb2[col];
#pragma unroll
    for (int v = 0; v < 8; ++v)
      sH[(half*8+v) * 132 + col] = acc[v] + bv;
  }
  __syncthreads();

  if (half == 0) {
    float* row = sH + m * 132;
    if (post_ln) {              // decoder: h = LN(gelu(h_new))
      float s = 0.f, ss = 0.f;
      for (int k = 0; k < C; ++k) {
        float g = gelu_f(row[k]);
        row[k] = g; s += g; ss += g * g;
      }
      float mu = s * (1.0f / C);
      float inv = rsqrtf(ss * (1.0f / C) - mu * mu + 1e-5f);
      for (int k = 0; k < C; ++k) {
        float o = (row[k] - mu) * inv * ln_g[k] + ln_b[k];
        h[node * C + k] = o;
        h16[node * C + k] = (f16)o;
      }
    } else {                    // encoder: h = h_new
      for (int k = 0; k < C; ++k) {
        float o = row[k];
        h[node * C + k] = o;
        h16[node * C + k] = (f16)o;
      }
    }
    if (do_coord) {
      coord[node*3+0] += x_agg[node*3+0];
      coord[node*3+1] += x_agg[node*3+1];
      coord[node*3+2] += x_agg[node*3+2];
    }
  }
}

// ---------------- readout ----------------
__global__ void out_kernel(const float* __restrict__ h, const float* __restrict__ w,
                           const float* __restrict__ b, float* __restrict__ out) {
  int i = blockIdx.x * blockDim.x + threadIdx.x;
  if (i >= NN) return;
  float a0 = b[0], a1 = b[1], a2 = b[2];
  const float* hr = h + (long)i * C;
#pragma unroll 4
  for (int k = 0; k < C; ++k) {
    float v = hr[k];
    a0 += v * w[k*3+0]; a1 += v * w[k*3+1]; a2 += v * w[k*3+2];
  }
  out[i*3+0] = a0; out[i*3+1] = a1; out[i*3+2] = a2;
}

// ---------------- host ----------------
extern "C" void kernel_launch(void* const* d_in, const int* in_sizes, int n_in,
                              void* d_out, int out_size, void* d_ws, size_t ws_size,
                              hipStream_t stream) {
  (void)in_sizes; (void)n_in; (void)out_size; (void)ws_size;
  const int*   src      = (const int*)d_in[0];
  const int*   dst      = (const int*)d_in[1];
  const float* nattr    = (const float*)d_in[2];
  const float* xyz      = (const float*)d_in[3];
  const float* emb_w1   = (const float*)d_in[4];
  const float* emb_b1   = (const float*)d_in[5];
  const float* ln1_g    = (const float*)d_in[6];
  const float* ln1_b    = (const float*)d_in[7];
  const float* emb_w2   = (const float*)d_in[8];
  const float* emb_b2   = (const float*)d_in[9];
  const float* ln2_g    = (const float*)d_in[10];
  const float* ln2_b    = (const float*)d_in[11];
  const float* e_w1     = (const float*)d_in[12];
  const float* e_b1     = (const float*)d_in[13];
  const float* e_w2     = (const float*)d_in[14];
  const float* e_b2     = (const float*)d_in[15];
  const float* n_w1     = (const float*)d_in[16];
  const float* n_b1     = (const float*)d_in[17];
  const float* n_w2     = (const float*)d_in[18];
  const float* n_b2     = (const float*)d_in[19];
  const float* c_w1     = (const float*)d_in[20];
  const float* c_b1     = (const float*)d_in[21];
  const float* c_w2     = (const float*)d_in[22];
  const float* dec_ln_g = (const float*)d_in[23];
  const float* dec_ln_b = (const float*)d_in[24];
  const float* out_w    = (const float*)d_in[25];
  const float* out_b    = (const float*)d_in[26];

  char* ws = (char*)d_ws;
  size_t off = 0;
  auto take = [&](size_t bytes) -> char* {
    char* p = ws + off;
    off += (bytes + 255) & ~(size_t)255;
    return p;
  };
  float* h     = (float*)take((size_t)NN * C * 4);
  f16*   h16   = (f16*)  take((size_t)NN * C * 2);
  float* h_agg = (float*)take((size_t)NN * C * 4);
  float* x_agg = (float*)take((size_t)NN * 3 * 4);
  float* coord = (float*)take((size_t)NN * 3 * 4);
  f16* emb_w1p = (f16*)take((size_t)FIN * C * 2);
  f16* emb_w2p = (f16*)take((size_t)C * C * 2);
  f16 *ew1p[3], *ew2p[3], *nw1p[3], *nw2p[3], *cw1p[3];
  for (int i = 0; i < 3; ++i) {
    ew1p[i] = (f16*)take((size_t)256 * C * 2);
    ew2p[i] = (f16*)take((size_t)C * C * 2);
    nw1p[i] = (f16*)take((size_t)256 * C * 2);
    nw2p[i] = (f16*)take((size_t)C * C * 2);
    cw1p[i] = (f16*)take((size_t)C * C * 2);
  }

  auto pack = [&](const float* s, f16* d, int K) {
    pack_b<<<(K * C) / 256, 256, 0, stream>>>(s, d, K);
  };
  pack(emb_w1, emb_w1p, FIN);
  pack(emb_w2, emb_w2p, C);
  for (int i = 0; i < 3; ++i) {
    pack(e_w1 + (size_t)i * 257 * C, ew1p[i], 256);  // rows 0..255 only
    pack(e_w2 + (size_t)i * C * C,   ew2p[i], C);
    pack(n_w1 + (size_t)i * 256 * C, nw1p[i], 256);
    pack(n_w2 + (size_t)i * C * C,   nw2p[i], C);
    pack(c_w1 + (size_t)i * C * C,   cw1p[i], C);
  }

  embed_kernel<<<NN / 16, 32, 0, stream>>>(nattr, emb_w1p, emb_b1, ln1_g, ln1_b,
                                           emb_w2p, emb_b2, ln2_g, ln2_b, h, h16);

  // decoder restarts coordinates from xyz
  hipMemcpyAsync(coord, xyz, (size_t)NN * 3 * 4, hipMemcpyDeviceToDevice, stream);

  for (int layer = 0; layer < 3; ++layer) {
    int dec = (layer > 0);
    hipMemsetAsync(h_agg, 0, (size_t)NN * C * 4, stream);
    if (dec) hipMemsetAsync(x_agg, 0, (size_t)NN * 3 * 4, stream);
    const float* xp = dec ? (const float*)coord : xyz;
    const float* ew1_last = e_w1 + (size_t)layer * 257 * C + 256 * C;  // k=256 row
    edge_kernel<<<NE / 32, 32, 0, stream>>>(
        src, dst, h16, xp,
        ew1p[layer], ew1_last, e_b1 + (size_t)layer * C,
        ew2p[layer], e_b2 + (size_t)layer * C,
        cw1p[layer], c_b1 + (size_t)layer * C, c_w2 + (size_t)layer * C,
        h_agg, x_agg, dec);
    node_kernel<<<NN / 16, 32, 0, stream>>>(
        h, h16, h_agg,
        nw1p[layer], n_b1 + (size_t)layer * C,
        nw2p[layer], n_b2 + (size_t)layer * C,
        dec_ln_g, dec_ln_b, coord, x_agg, dec, dec);
  }

  out_kernel<<<(NN + 255) / 256, 256, 0, stream>>>(h, out_w, out_b, (float*)d_out);
}
